// SANet_69441031242154
// MI455X (gfx1250) — compile-verified, matrix operand order checked by reading, not run on hardware
//
#include <hip/hip_runtime.h>

// ---------------------------------------------------------------------------
// CDNA5 (gfx1250) SANet self-attention pipeline.
//  - All GEMMs on v_wmma_f32_16x16x32_bf16 (wave32).
//  - Attention key-panels staged into LDS via TDM (tensor_load_to_lds),
//    double buffered, padded rows (D# pad fields) to avoid bank conflicts.
// ---------------------------------------------------------------------------

typedef __bf16 bf16;
typedef __attribute__((ext_vector_type(16))) __bf16 v16bf;
typedef __attribute__((ext_vector_type(8)))  float  v8f;
typedef __attribute__((ext_vector_type(4)))  unsigned int u32x4;
typedef __attribute__((ext_vector_type(8)))  int i32x8;
typedef __attribute__((ext_vector_type(4)))  int i32x4;

#if defined(__has_builtin)
#if __has_builtin(__builtin_amdgcn_tensor_load_to_lds) && \
    __has_builtin(__builtin_amdgcn_s_wait_tensorcnt)
#define HAVE_TDM 1
#endif
#endif
#ifndef HAVE_TDM
#define HAVE_TDM 0
#endif

union FragU { v16bf v; uint4 q[2]; };

// A-operand fragment (16x32 bf16) from row-major [rows, ld] at (row0, k0).
// Identical layout serves the B operand when B is stored as [N, K]
// (B^T row-major): lane = N column, K = 8h+s (s<8), 16+8h+s-8 (s>=8), h = lane/16.
__device__ __forceinline__ v16bf load_frag(const bf16* __restrict__ base,
                                           int ld, int row, int k0, int lane)
{
    int r  = row + (lane & 15);
    int kk = k0  + ((lane >> 4) << 3);
    const bf16* p = base + (size_t)r * (size_t)ld + kk;
    FragU u;
    u.q[0] = *reinterpret_cast<const uint4*>(p);
    u.q[1] = *reinterpret_cast<const uint4*>(p + 16);
    return u.v;
}

__device__ __forceinline__ v8f wmma_bf(v16bf a, v16bf b, v8f c)
{
    return __builtin_amdgcn_wmma_f32_16x16x32_bf16(false, a, false, b,
                                                   (short)0, c, false, false);
}

__device__ __forceinline__ v8f vzero8()
{
    v8f z = {0.f, 0.f, 0.f, 0.f, 0.f, 0.f, 0.f, 0.f};
    return z;
}

#if HAVE_TDM
// ---------------------------------------------------------------------------
// TDM: async-load a [64 x C] bf16 tile (row-major, row stride C) from global
// into LDS at byte offset lds_off, padding the LDS destination by 4 DWORDs
// after every 256 DWORDs (i.e. per 512-element row) -> LDS pitch = C+8 elems.
// D# layout per CDNA5 ISA ch.8 (group0 128b, group1 256b; groups 2/3 zero
// for a 2D tile).
// ---------------------------------------------------------------------------
__device__ __forceinline__ void tdm_load_tile(const bf16* gsrc,
                                              unsigned lds_off,
                                              int C, int Npix)
{
    unsigned long long ga = (unsigned long long)(size_t)gsrc;
    u32x4 g0;
    g0[0] = 1u;                                   // count=1, user descriptor
    g0[1] = lds_off;                              // lds_addr (bytes)
    g0[2] = (unsigned)(ga & 0xFFFFFFFFull);       // global_addr[31:0]
    g0[3] = (unsigned)((ga >> 32) & 0x01FFFFFFull) | 0x80000000u; // addr[56:32] | type=2

    i32x8 g1;
    // data_size=2B, pad_enable, pad_interval=7 (256 dw), pad_amount=3 (4 dw)
    g1[0] = (int)((1u << 16) | (1u << 20) | (7u << 22) | (3u << 25));
    g1[1] = (int)(((unsigned)C & 0xFFFFu) << 16);                   // tensor_dim0 lo
    g1[2] = (int)(((unsigned)C >> 16) |
                  (((unsigned)Npix & 0xFFFFu) << 16));              // dim0 hi | tensor_dim1 lo
    g1[3] = (int)(((unsigned)Npix >> 16) |
                  (((unsigned)C & 0xFFFFu) << 16));                 // dim1 hi | tile_dim0=C
    g1[4] = 64;                                                     // tile_dim1 = 64 rows
    g1[5] = C;                                                      // tensor_dim0_stride
    g1[6] = 0;
    g1[7] = 0;

    i32x4 z4 = {0, 0, 0, 0};
#if defined(__clang_major__) && (__clang_major__ >= 23)
    i32x8 z8 = {0, 0, 0, 0, 0, 0, 0, 0};
    __builtin_amdgcn_tensor_load_to_lds(g0, g1, z4, z4, z8, 0);
#else
    __builtin_amdgcn_tensor_load_to_lds(g0, g1, z4, z4, 0);
#endif
}
#endif  // HAVE_TDM

// ---------------------------------------------------------------------------
// Weight fp32 -> bf16 conversion
// ---------------------------------------------------------------------------
__global__ __launch_bounds__(256) void w_to_bf16_kernel(const float* __restrict__ src,
                                                        bf16* __restrict__ dst, int n)
{
    int i = blockIdx.x * 256 + threadIdx.x;
    if (i < n) dst[i] = (bf16)src[i];
}

// ---------------------------------------------------------------------------
// channel_norm: per (b,c) row over N=4096 spatial, unbiased var (ddof=1).
// Writes transposed bf16 output X^T [B, N, C].
// ---------------------------------------------------------------------------
__global__ __launch_bounds__(256) void norm_kernel(const float* __restrict__ src,
                                                   bf16* __restrict__ dstT,
                                                   int C, int Npix)
{
    const int NIT = 4096 / 256;
    int b = blockIdx.x / C;
    int c = blockIdx.x - b * C;
    const float* row = src + (size_t)blockIdx.x * Npix;

    float vals[NIT];
    float s1 = 0.f, s2 = 0.f;
#pragma unroll
    for (int i = 0; i < NIT; ++i) {
        float v = row[threadIdx.x + i * 256];
        vals[i] = v;
        s1 += v;
        s2 += v * v;
    }
#pragma unroll
    for (int m = 1; m < 32; m <<= 1) {
        s1 += __shfl_xor(s1, m);
        s2 += __shfl_xor(s2, m);
    }
    __shared__ float sh1[8], sh2[8];
    int wave = threadIdx.x >> 5, lane = threadIdx.x & 31;
    if (lane == 0) { sh1[wave] = s1; sh2[wave] = s2; }
    __syncthreads();
    float t1 = 0.f, t2 = 0.f;
#pragma unroll
    for (int w = 0; w < 8; ++w) { t1 += sh1[w]; t2 += sh2[w]; }

    float mean = t1 / (float)Npix;
    float var  = (t2 - t1 * mean) / (float)(Npix - 1);
    float inv  = rsqrtf(var + 1e-5f);

    bf16* dcol = dstT + (size_t)b * Npix * C + c;
#pragma unroll
    for (int i = 0; i < NIT; ++i) {
        int n = threadIdx.x + i * 256;
        dcol[(size_t)n * C] = (bf16)((vals[i] - mean) * inv);
    }
}

// ---------------------------------------------------------------------------
// Generic WMMA GEMM:  D[M,N] = A[M,K] (row major) x Bt[N,K] (B^T row major)
// + bias (0:none, 1:per-n, 2:per-m), optional fp32 output with residual.
// Block = 256 thr = 8 waves in 4(M)x2(N); wave tile 64x64; block tile 256x128.
// ---------------------------------------------------------------------------
__global__ __launch_bounds__(256) void gemm_bf16_kernel(
    const bf16* __restrict__ A, const bf16* __restrict__ Bt,
    const float* __restrict__ bias, int bias_mode,
    bf16* __restrict__ Dbf, float* __restrict__ Df,
    const float* __restrict__ Res,
    int M, int N, int K,
    unsigned long long sA, unsigned long long sB,
    unsigned long long sD, unsigned long long sR)
{
    int lane = threadIdx.x & 31;
    int wave = threadIdx.x >> 5;
    int b    = blockIdx.z;
    A  += (size_t)b * sA;
    Bt += (size_t)b * sB;

    int m0 = blockIdx.x * 256 + (wave >> 1) * 64;
    int n0 = blockIdx.y * 128 + (wave & 1) * 64;

    v8f acc[4][4];
#pragma unroll
    for (int i = 0; i < 4; ++i)
#pragma unroll
        for (int j = 0; j < 4; ++j) acc[i][j] = vzero8();

    for (int k0 = 0; k0 < K; k0 += 32) {
        if (k0 + 32 < K) {  // -> global_prefetch_b8
            __builtin_prefetch(A  + (size_t)(m0 + (lane & 15)) * K + k0 + 32, 0, 1);
            __builtin_prefetch(Bt + (size_t)(n0 + (lane & 15)) * K + k0 + 32, 0, 1);
        }
        v16bf af[4], bfg[4];
#pragma unroll
        for (int i = 0; i < 4; ++i) af[i]  = load_frag(A,  K, m0 + 16 * i, k0, lane);
#pragma unroll
        for (int j = 0; j < 4; ++j) bfg[j] = load_frag(Bt, K, n0 + 16 * j, k0, lane);
#pragma unroll
        for (int i = 0; i < 4; ++i)
#pragma unroll
            for (int j = 0; j < 4; ++j)
                acc[i][j] = wmma_bf(af[i], bfg[j], acc[i][j]);
    }

    int nl = lane & 15, half = lane >> 4;
    size_t dbase = (size_t)b * sD;
#pragma unroll
    for (int i = 0; i < 4; ++i) {
#pragma unroll
        for (int j = 0; j < 4; ++j) {
            int mt = m0 + 16 * i + 8 * half;
            int nt = n0 + 16 * j + nl;
            float bn = (bias_mode == 1) ? bias[nt] : 0.f;
#pragma unroll
            for (int r = 0; r < 8; ++r) {
                float v = acc[i][j][r] + bn;
                if (bias_mode == 2) v += bias[mt + r];
                size_t off = (size_t)(mt + r) * N + nt;
                if (Df) {
                    float o = v;
                    if (Res) o += Res[(size_t)b * sR + off];
                    Df[dbase + off] = o;
                } else {
                    Dbf[dbase + off] = (bf16)v;
                }
            }
        }
    }
}

// ---------------------------------------------------------------------------
// Attention S-tile compute shared by both passes: key panel staged in LDS.
// LDS pitch = C + 8 bf16 (TDM pad: 4 DWORDs per 256-DWORD row) so the 16
// row-parallel ds_load_b128 of a B-fragment hit 64 distinct banks.
// ---------------------------------------------------------------------------
#define ATT_PITCH(Cv) ((Cv) + 8)

__device__ __forceinline__ void attn_s_tile(const bf16* __restrict__ fq,
                                            const bf16* __restrict__ gt_lds,
                                            int i0, int C, int pitch, int lane,
                                            v8f acc[4])
{
    for (int k0 = 0; k0 < C; k0 += 32) {
        v16bf aq = load_frag(fq, C, i0, k0, lane);
#pragma unroll
        for (int t = 0; t < 4; ++t) {
            v16bf bg = load_frag(gt_lds, pitch, 16 * t, k0, lane);
            acc[t] = wmma_bf(aq, bg, acc[t]);
        }
    }
}

// Stage control: returns pointer to the LDS tile for this iteration.
// TDM path: wave0 waits TENSORcnt for tile j0 (prefetched last iter), block
// barrier, wave0 launches the next tile into the other buffer.
// Fallback: cooperative global->LDS copy with ds_store_b128.
__device__ __forceinline__ const bf16* attn_stage(bf16* smem, const bf16* gk,
                                                  int j0, int Npix, int C,
                                                  int wave, int* cur)
{
    const int pitch = ATT_PITCH(C);
    const unsigned bufBytes = (unsigned)(64u * (unsigned)pitch * sizeof(bf16));
#if HAVE_TDM
    if (wave == 0) __builtin_amdgcn_s_wait_tensorcnt(0);
    __syncthreads();
    if (wave == 0 && (j0 + 64) < Npix)
        tdm_load_tile(gk + (size_t)(j0 + 64) * C,
                      (unsigned)((*cur ^ 1)) * bufBytes, C, Npix);
    const bf16* gt = smem + (size_t)(*cur) * 64 * pitch;
    *cur ^= 1;
    return gt;
#else
    (void)bufBytes; (void)wave; (void)cur;
    __syncthreads();
    const int vecs = C / 8;  // uint4 chunks per row
    for (int idx = threadIdx.x; idx < 64 * vecs; idx += 256) {
        int rr = idx / vecs, cc = (idx - rr * vecs) * 8;
        *reinterpret_cast<uint4*>(&smem[(size_t)rr * pitch + cc]) =
            *reinterpret_cast<const uint4*>(&gk[(size_t)(j0 + rr) * C + cc]);
    }
    __syncthreads();
    return smem;
#endif
}

// ---------------------------------------------------------------------------
// Attention pass 1: per-row online softmax stats (max m_i, sum l_i).
// One wave owns 16 query rows; grid = Npix/128 blocks of 256 threads.
// ---------------------------------------------------------------------------
__global__ __launch_bounds__(256) void attn_stats_kernel(
    const bf16* __restrict__ fq, const bf16* __restrict__ gk,
    float* __restrict__ m_out, float* __restrict__ l_out,
    int Npix, int C)
{
    extern __shared__ bf16 smem[];  // 2 x 64 x (C+8) bf16
    int lane = threadIdx.x & 31;
    int wave = threadIdx.x >> 5;
    int i0   = (blockIdx.x * 8 + wave) * 16;
    int cur  = 0;

#if HAVE_TDM
    if (wave == 0) tdm_load_tile(gk, 0u, C, Npix);  // prologue: tile 0 -> buf0
#endif

    float rm[8], rl[8];
#pragma unroll
    for (int r = 0; r < 8; ++r) { rm[r] = -3.0e38f; rl[r] = 0.f; }

    for (int j0 = 0; j0 < Npix; j0 += 64) {
        const bf16* gt = attn_stage(smem, gk, j0, Npix, C, wave, &cur);
        v8f acc[4];
#pragma unroll
        for (int t = 0; t < 4; ++t) acc[t] = vzero8();
        attn_s_tile(fq, gt, i0, C, ATT_PITCH(C), lane, acc);

#pragma unroll
        for (int r = 0; r < 8; ++r) {
            float tm = fmaxf(fmaxf(acc[0][r], acc[1][r]), fmaxf(acc[2][r], acc[3][r]));
#pragma unroll
            for (int m = 1; m < 16; m <<= 1) tm = fmaxf(tm, __shfl_xor(tm, m));
            float nm = fmaxf(rm[r], tm);
            float ts = __expf(acc[0][r] - nm) + __expf(acc[1][r] - nm) +
                       __expf(acc[2][r] - nm) + __expf(acc[3][r] - nm);
#pragma unroll
            for (int m = 1; m < 16; m <<= 1) ts += __shfl_xor(ts, m);
            rl[r] = rl[r] * __expf(rm[r] - nm) + ts;
            rm[r] = nm;
        }
    }
    if ((lane & 15) == 0) {
        int half = lane >> 4;
#pragma unroll
        for (int r = 0; r < 8; ++r) {
            int row = i0 + r + 8 * half;
            m_out[row] = rm[r];
            l_out[row] = rl[r];
        }
    }
}

// ---------------------------------------------------------------------------
// Attention pass 2: recompute S tiles, P = exp(s - m)/l, store bf16 P [N,N].
// ---------------------------------------------------------------------------
__global__ __launch_bounds__(256) void attn_p_kernel(
    const bf16* __restrict__ fq, const bf16* __restrict__ gk,
    const float* __restrict__ m_in, const float* __restrict__ l_in,
    bf16* __restrict__ P, int Npix, int C)
{
    extern __shared__ bf16 smem[];
    int lane = threadIdx.x & 31;
    int wave = threadIdx.x >> 5;
    int half = lane >> 4;
    int nl   = lane & 15;
    int i0   = (blockIdx.x * 8 + wave) * 16;
    int cur  = 0;

#if HAVE_TDM
    if (wave == 0) tdm_load_tile(gk, 0u, C, Npix);
#endif

    float mv[8], li[8];
#pragma unroll
    for (int r = 0; r < 8; ++r) {
        int row = i0 + r + 8 * half;
        mv[r] = m_in[row];
        li[r] = 1.f / l_in[row];
    }

    for (int j0 = 0; j0 < Npix; j0 += 64) {
        const bf16* gt = attn_stage(smem, gk, j0, Npix, C, wave, &cur);
        v8f acc[4];
#pragma unroll
        for (int t = 0; t < 4; ++t) acc[t] = vzero8();
        attn_s_tile(fq, gt, i0, C, ATT_PITCH(C), lane, acc);

#pragma unroll
        for (int r = 0; r < 8; ++r) {
            size_t rowoff = (size_t)(i0 + r + 8 * half) * Npix;
#pragma unroll
            for (int t = 0; t < 4; ++t) {
                float p = __expf(acc[t][r] - mv[r]) * li[r];
                P[rowoff + j0 + 16 * t + nl] = (bf16)p;
            }
        }
    }
}

// ---------------------------------------------------------------------------
// Host-side orchestration
// ---------------------------------------------------------------------------
extern "C" void kernel_launch(void* const* d_in, const int* in_sizes, int n_in,
                              void* d_out, int out_size, void* d_ws, size_t ws_size,
                              hipStream_t stream)
{
    (void)in_sizes; (void)n_in; (void)out_size; (void)ws_size;

    const int B = 4, C = 512, Npix = 4096;
    const size_t MB = 1ull << 20;

    const float* f_c   = (const float*)d_in[0];
    const float* f_s   = (const float*)d_in[1];
    const float* W_c1  = (const float*)d_in[2];
    const float* b_c1  = (const float*)d_in[3];
    const float* W_s1  = (const float*)d_in[4];
    const float* b_s1  = (const float*)d_in[5];
    const float* W_s2  = (const float*)d_in[6];
    const float* b_s2  = (const float*)d_in[7];
    const float* W_csc = (const float*)d_in[8];
    const float* b_csc = (const float*)d_in[9];
    float* out = (float*)d_out;

    char* ws = (char*)d_ws;
    bf16* xcT  = (bf16*)(ws +  0 * MB);   // X̂_c^T (aliased by P later)
    bf16* xsT  = (bf16*)(ws + 16 * MB);   // X̂_s^T (aliased by P later)
    bf16* fQ   = (bf16*)(ws + 32 * MB);   // f  [B,N,C]
    bf16* gK   = (bf16*)(ws + 48 * MB);   // g  [B,N,C]
    bf16* hV   = (bf16*)(ws + 64 * MB);   // h  [B,C,N]
    bf16* OT   = (bf16*)(ws + 80 * MB);   // O^T [B,N,C]
    bf16* wc1b = (bf16*)(ws + 96 * MB);
    bf16* ws1b = wc1b + (size_t)C * C;
    bf16* ws2b = ws1b + (size_t)C * C;
    bf16* wcsb = ws2b + (size_t)C * C;
    float* statM = (float*)(ws + 98 * MB);
    float* statL = statM + Npix;
    bf16* P = (bf16*)(ws);                // 32 MB, reused per batch

    const unsigned long long sNC = (unsigned long long)Npix * C;
    const int WN = C * C;
    const size_t attn_lds = 2u * 64u * (size_t)ATT_PITCH(C) * sizeof(bf16);

    // 1) weights -> bf16
    w_to_bf16_kernel<<<WN / 256, 256, 0, stream>>>(W_c1,  wc1b, WN);
    w_to_bf16_kernel<<<WN / 256, 256, 0, stream>>>(W_s1,  ws1b, WN);
    w_to_bf16_kernel<<<WN / 256, 256, 0, stream>>>(W_s2,  ws2b, WN);
    w_to_bf16_kernel<<<WN / 256, 256, 0, stream>>>(W_csc, wcsb, WN);

    // 2) channel_norm -> transposed bf16
    norm_kernel<<<B * C, 256, 0, stream>>>(f_c, xcT, C, Npix);
    norm_kernel<<<B * C, 256, 0, stream>>>(f_s, xsT, C, Npix);

    // 3) f = X̂_c^T · W_c1^T ; 4) g = X̂_s^T · W_s1^T   -> [B,N,C]
    gemm_bf16_kernel<<<dim3(16, 4, B), 256, 0, stream>>>(
        xcT, wc1b, b_c1, 1, fQ, nullptr, nullptr,
        Npix, C, C, sNC, 0ull, sNC, 0ull);
    gemm_bf16_kernel<<<dim3(16, 4, B), 256, 0, stream>>>(
        xsT, ws1b, b_s1, 1, gK, nullptr, nullptr,
        Npix, C, C, sNC, 0ull, sNC, 0ull);
    // 5) h = W_s2 · X̂_s  -> [B,C,N]
    gemm_bf16_kernel<<<dim3(2, 32, B), 256, 0, stream>>>(
        ws2b, xsT, b_s2, 2, hV, nullptr, nullptr,
        C, Npix, C, 0ull, sNC, sNC, 0ull);

    // 6) attention per batch (P reuses dead X̂ buffers)
    for (int b = 0; b < B; ++b) {
        const bf16* fb = fQ + (size_t)b * sNC;
        const bf16* gb = gK + (size_t)b * sNC;
        const bf16* hb = hV + (size_t)b * sNC;
        bf16* ob = OT + (size_t)b * sNC;

        attn_stats_kernel<<<Npix / 128, 256, attn_lds, stream>>>(
            fb, gb, statM, statL, Npix, C);
        attn_p_kernel<<<Npix / 128, 256, attn_lds, stream>>>(
            fb, gb, statM, statL, P, Npix, C);
        gemm_bf16_kernel<<<dim3(16, 4, 1), 256, 0, stream>>>(
            P, hb, nullptr, 0, ob, nullptr, nullptr,
            Npix, C, Npix, 0ull, 0ull, 0ull, 0ull);
    }

    // 7) result = W_csc · O + b_csc + f_c  (fp32, [B,C,H,W])
    gemm_bf16_kernel<<<dim3(2, 32, B), 256, 0, stream>>>(
        wcsb, OT, b_csc, 2, nullptr, out, f_c,
        C, Npix, C, 0ull, sNC, sNC, sNC);
}
